// GQA_15771119911030
// MI455X (gfx1250) — matrix operand order repaired and checked by reading, not compiled, not run on hardware
//
#include <hip/hip_runtime.h>
#include <hip/hip_bf16.h>
#include <math.h>

// ---------------------------------------------------------------------------
// GQA forward for MI455X (gfx1250, wave32, WMMA).
// One streaming fp32->bf16 conversion pass, then every GEMM / attention inner
// loop runs on pure bf16 operands feeding v_wmma_f32_16x16x32_bf16 (fp32 acc).
// bf16 weights (~80MB) fit in the 192MB L2, so per-tile re-reads stay on chip.
// ---------------------------------------------------------------------------

typedef __attribute__((ext_vector_type(16))) __bf16 v16bf;
typedef __attribute__((ext_vector_type(8)))  __bf16 v8bf;
typedef __attribute__((ext_vector_type(8)))  float  v8f;

constexpr int B   = 2;
constexpr int S   = 2048;
constexpr int D   = 4096;
constexpr int NH  = 32;
constexpr int NKV = 8;
constexpr int HD  = 128;
constexpr int NREP = NH / NKV;

// Concatenate two 8-element bf16 chunks into one 16-element WMMA fragment.
__device__ __forceinline__ v16bf cat16(const __bf16* p0, const __bf16* p1) {
  v8bf lo = *(const v8bf*)p0;
  v8bf hi = *(const v8bf*)p1;
  return __builtin_shufflevector(lo, hi, 0, 1, 2, 3, 4, 5, 6, 7, 8, 9, 10, 11,
                                 12, 13, 14, 15);
}

// ---------------------------------------------------------------------------
// Streaming fp32 -> bf16 conversion (8 elements / thread, b128 in, b128 out).
// ---------------------------------------------------------------------------
__global__ void __launch_bounds__(256)
cvt_f32_to_bf16(const float* __restrict__ in, __bf16* __restrict__ out,
                long long n) {
  const long long i =
      ((long long)blockIdx.x * blockDim.x + threadIdx.x) * 8;
  if (i >= n) return;
  const float4 a = *(const float4*)(in + i);
  const float4 b = *(const float4*)(in + i + 4);
  v8bf o;
  o[0] = (__bf16)a.x; o[1] = (__bf16)a.y; o[2] = (__bf16)a.z; o[3] = (__bf16)a.w;
  o[4] = (__bf16)b.x; o[5] = (__bf16)b.y; o[6] = (__bf16)b.z; o[7] = (__bf16)b.w;
  *(v8bf*)(out + i) = o;
}

// ---------------------------------------------------------------------------
// C[M,N] = A[M,K] * W[N,K]^T, A/W row-major bf16, contraction along K.
// One wave computes 16(M) x 128(N): 1 A-frag feeds 8 WMMAs per 32-deep k-step.
// OutT selects fp32 (final output) or bf16 (intermediates) stores.
// ---------------------------------------------------------------------------
template <typename OutT>
__global__ void __launch_bounds__(256)
gemm_wmma_bf16(const __bf16* __restrict__ A, const __bf16* __restrict__ W,
               OutT* __restrict__ C, int M, int N, int K) {
  const int ntiles = N >> 7;
  const int wave = (int)((blockIdx.x * blockDim.x + threadIdx.x) >> 5);
  const int lane = (int)(threadIdx.x & 31);
  const int mt = wave / ntiles;
  const int nt = wave - mt * ntiles;
  if (mt * 16 >= M) return;

  const int ln = lane & 15;  // row (A) / col (B) within 16-wide tile
  const int hf = lane >> 4;  // half-wave selects K sub-chunks
  const size_t arow = (size_t)(mt * 16 + ln) * (size_t)K;

  v8f acc[8] = {};

  for (int k0 = 0; k0 < K; k0 += 32) {
    // A fragment 16x32: lanes 0-15 K {0..7,16..23}; lanes 16-31 {8..15,24..31}
    const __bf16* ap = A + arow + k0 + hf * 8;
    if (k0 + 32 < K) __builtin_prefetch(ap + 32, 0, 1);  // global_prefetch_b8
    const v16bf a = cat16(ap, ap + 16);
#pragma unroll
    for (int t = 0; t < 8; ++t) {
      // B fragment 32x16: column = W row, lanes 0-15 K {0..7,16..23}; 16-31 {8..15,24..31}.
      const __bf16* wp =
          W + (size_t)(nt * 128 + t * 16 + ln) * (size_t)K + k0 + hf * 8;
      const v16bf b = cat16(wp, wp + 16);
      acc[t] = __builtin_amdgcn_wmma_f32_16x16x32_bf16(
          false, a, false, b, (short)0, acc[t], false, false);
    }
  }

  // C/D layout: VGPR j -> row j (lanes 0-15) or row j+8 (lanes 16-31)
#pragma unroll
  for (int t = 0; t < 8; ++t) {
    OutT* cp =
        C + (size_t)(mt * 16 + hf * 8) * (size_t)N + nt * 128 + t * 16 + ln;
#pragma unroll
    for (int j = 0; j < 8; ++j) cp[(size_t)j * N] = (OutT)acc[t][j];
  }
}

// ---------------------------------------------------------------------------
// RoPE in place on bf16 Q [B,S,NH,HD] and K [B,S,NKV,HD] (fp32 math).
// start_pos == 0, so position == s.
// ---------------------------------------------------------------------------
__global__ void __launch_bounds__(256)
rope_kernel(__bf16* __restrict__ q, __bf16* __restrict__ k,
            const float* __restrict__ fc, const float* __restrict__ fs) {
  const int pairs = HD / 2;
  const int per_tok = (NH + NKV) * pairs;
  const long long total = (long long)B * S * per_tok;
  const long long i = (long long)blockIdx.x * blockDim.x + threadIdx.x;
  if (i >= total) return;
  const long long tok = i / per_tok;
  int rem = (int)(i - tok * per_tok);
  const int h = rem / pairs;
  const int p = rem - h * pairs;
  const int s = (int)(tok % S);
  const float c  = fc[s * pairs + p];
  const float sn = fs[s * pairs + p];
  __bf16* base = (h < NH) ? (q + (tok * NH + h) * HD)
                          : (k + (tok * NKV + (h - NH)) * HD);
  const float xe = (float)base[2 * p];
  const float xo = (float)base[2 * p + 1];
  base[2 * p]     = (__bf16)(xe * c - xo * sn);
  base[2 * p + 1] = (__bf16)(xe * sn + xo * c);
}

// ---------------------------------------------------------------------------
// Causal flash attention (bf16 operands, fp32 softmax/accum).
// One wave per (batch, head, 16-query tile), 32 keys per iteration so P is
// exactly a 16x32 WMMA A-operand. Softmax state lives in the C-layout: per
// lane, index j belongs to row j + 8*(lane>=16); row reductions are 16-lane
// shfl_xor within half-waves. P transposes through 1KB/wave of LDS guarded by
// an explicit s_wait_dscnt (CDNA5 split counter).
// ---------------------------------------------------------------------------
__global__ void __launch_bounds__(128)
attn_fwd_wmma(const __bf16* __restrict__ Q, const __bf16* __restrict__ Kc,
              const __bf16* __restrict__ Vc, __bf16* __restrict__ O) {
  __shared__ __bf16 pbuf[4][16][32];  // per-wave P staging (row-major 16x32)

  const int wib  = threadIdx.x >> 5;
  const int lane = threadIdx.x & 31;
  const int ln = lane & 15;
  const int hf = lane >> 4;
  const int QT = S / 16;
  const int wave = blockIdx.x * 4 + wib;
  const int b  = wave / (NH * QT);
  int r        = wave - b * NH * QT;
  const int h  = r / QT;
  const int qt = r - h * QT;
  const int kvh = h / NREP;
  const int qbase = qt * 16;
  const float scale = 0.08838834764831845f;  // 1/sqrt(HD), applied to scores

  // Preload the 4 Q A-fragments covering HD=128 (constant over the kv loop).
  v16bf qa[4];
  {
    const __bf16* qp = Q + ((size_t)(b * S + qbase + ln) * NH + h) * HD;
#pragma unroll
    for (int kk = 0; kk < 4; ++kk) {
      const __bf16* p = qp + kk * 32 + hf * 8;
      qa[kk] = cat16(p, p + 16);
    }
  }

  float mrow[8], lrow[8];
  v8f acc[8] = {};
#pragma unroll
  for (int j = 0; j < 8; ++j) { mrow[j] = -__builtin_inff(); lrow[j] = 0.0f; }

  for (int kb = 0; kb < qbase + 16; kb += 32) {
    // ---- scores: Q[16,128] x K[32,128]^T -> two 16x16 tiles --------------
    v8f s0 = {}, s1 = {};
#pragma unroll
    for (int kk = 0; kk < 4; ++kk) {
      const __bf16* kp0 =
          Kc + ((size_t)(b * S + kb + ln) * NKV + kvh) * HD + kk * 32 + hf * 8;
      const __bf16* kp1 = kp0 + (size_t)16 * NKV * HD;
      const v16bf b0 = cat16(kp0, kp0 + 16);
      const v16bf b1 = cat16(kp1, kp1 + 16);
      s0 = __builtin_amdgcn_wmma_f32_16x16x32_bf16(
          false, qa[kk], false, b0, (short)0, s0, false, false);
      s1 = __builtin_amdgcn_wmma_f32_16x16x32_bf16(
          false, qa[kk], false, b1, (short)0, s1, false, false);
    }

    // ---- causal mask + online softmax (per-row state in C-layout) --------
    const int key0 = kb + ln;
    const int key1 = key0 + 16;
    float alpha[8];
#pragma unroll
    for (int j = 0; j < 8; ++j) {
      const int rowq = qbase + j + hf * 8;
      float v0 = (key0 <= rowq) ? s0[j] * scale : -__builtin_inff();
      float v1 = (key1 <= rowq) ? s1[j] * scale : -__builtin_inff();
      float mb = fmaxf(v0, v1);
#pragma unroll
      for (int off = 8; off > 0; off >>= 1)
        mb = fmaxf(mb, __shfl_xor(mb, off, 16));
      const float mn = fmaxf(mrow[j], mb);
      alpha[j] = __expf(mrow[j] - mn);
      mrow[j] = mn;
      const float p0 = __expf(v0 - mn);
      const float p1 = __expf(v1 - mn);
      pbuf[wib][j + hf * 8][ln]      = (__bf16)p0;
      pbuf[wib][j + hf * 8][16 + ln] = (__bf16)p1;
      float rs = p0 + p1;
#pragma unroll
      for (int off = 8; off > 0; off >>= 1) rs += __shfl_xor(rs, off, 16);
      lrow[j] = lrow[j] * alpha[j] + rs;
    }

    // LDS stores from this wave are in order; wait before the transposed read.
    __asm__ volatile("s_wait_dscnt 0" ::: "memory");

    // ---- P as A-fragment (16x32) ----------------------------------------
    const v16bf pf =
        cat16(&pbuf[wib][ln][hf * 16], &pbuf[wib][ln][hf * 16 + 8]);

    // ---- O = diag(alpha)*O + P * V[32,128] ------------------------------
#pragma unroll
    for (int nt = 0; nt < 8; ++nt) {
      const __bf16* vp =
          Vc + ((size_t)(b * S + kb + hf * 16) * NKV + kvh) * HD + nt * 16 + ln;
      v16bf vf;
#pragma unroll
      for (int i = 0; i < 16; ++i) vf[i] = vp[(size_t)i * NKV * HD];
#pragma unroll
      for (int j = 0; j < 8; ++j) acc[nt][j] *= alpha[j];
      acc[nt] = __builtin_amdgcn_wmma_f32_16x16x32_bf16(
          false, pf, false, vf, (short)0, acc[nt], false, false);
    }
  }

  // ---- epilogue: O /= l, write bf16 [B,S,NH*HD] for the out-projection ----
#pragma unroll
  for (int j = 0; j < 8; ++j) {
    const float inv = 1.0f / lrow[j];
    const int rowq = qbase + j + hf * 8;
    __bf16* op = O + (size_t)(b * S + rowq) * (NH * HD) + h * HD;
#pragma unroll
    for (int nt = 0; nt < 8; ++nt)
      op[nt * 16 + ln] = (__bf16)(acc[nt][j] * inv);
  }
}

// ---------------------------------------------------------------------------
extern "C" void kernel_launch(void* const* d_in, const int* in_sizes, int n_in,
                              void* d_out, int out_size, void* d_ws,
                              size_t ws_size, hipStream_t stream) {
  const float* x  = (const float*)d_in[0];
  const float* wq = (const float*)d_in[1];
  const float* wk = (const float*)d_in[2];
  const float* wv = (const float*)d_in[3];
  const float* wo = (const float*)d_in[4];
  const float* fc = (const float*)d_in[5];
  const float* fs = (const float*)d_in[6];
  // d_in[7]=mask (causality computed in-kernel), d_in[8..9]=zero caches
  // (start_pos==0, kv comes entirely from this step), d_in[10]=start_pos(0).
  float* out = (float*)d_out;

  // bf16 workspace layout
  const long long nx  = (long long)B * S * D;        // 16M
  const long long nwq = (long long)NH * HD * D;      // 16M
  const long long nwk = (long long)NKV * HD * D;     // 4M
  const long long nq  = (long long)B * S * NH * HD;  // 16M
  const long long nkv = (long long)B * S * NKV * HD; // 4M

  __bf16* p = (__bf16*)d_ws;
  __bf16* xbf  = p; p += nx;
  __bf16* wqbf = p; p += nwq;
  __bf16* wkbf = p; p += nwk;
  __bf16* wvbf = p; p += nwk;
  __bf16* wobf = p; p += nwq;
  __bf16* qbf  = p; p += nq;
  __bf16* kbf  = p; p += nkv;
  __bf16* vbf  = p; p += nkv;
  __bf16* abf  = p; p += nq;

  auto cvt = [&](const float* src, __bf16* dst, long long n) {
    const int blocks = (int)((n / 8 + 255) / 256);
    cvt_f32_to_bf16<<<blocks, 256, 0, stream>>>(src, dst, n);
  };
  cvt(x, xbf, nx);
  cvt(wq, wqbf, nwq);
  cvt(wk, wkbf, nwk);
  cvt(wv, wvbf, nwk);
  cvt(wo, wobf, nwq);

  const int M = B * S;
  // QKV projections -> bf16 intermediates
  {
    auto g = [&](const __bf16* W, __bf16* C, int N) {
      const int waves = (M / 16) * (N / 128);
      gemm_wmma_bf16<__bf16>
          <<<(waves * 32 + 255) / 256, 256, 0, stream>>>(xbf, W, C, M, N, D);
    };
    g(wqbf, qbf, NH * HD);
    g(wkbf, kbf, NKV * HD);
    g(wvbf, vbf, NKV * HD);
  }

  // RoPE on Q and K
  {
    const long long total = (long long)B * S * (NH + NKV) * (HD / 2);
    rope_kernel<<<(int)((total + 255) / 256), 256, 0, stream>>>(qbf, kbf, fc,
                                                                fs);
  }

  // causal GQA flash attention -> bf16 attention output
  {
    const int waves = B * NH * (S / 16);  // 8192
    attn_fwd_wmma<<<waves / 4, 128, 0, stream>>>(qbf, kbf, vbf, abf);
  }

  // output projection -> fp32 result
  {
    const int waves = (M / 16) * (D / 128);
    gemm_wmma_bf16<float>
        <<<(waves * 32 + 255) / 256, 256, 0, stream>>>(abf, wobf, out, M, D,
                                                       NH * HD);
  }
}
